// SizeGNN_39805756899773
// MI455X (gfx1250) — matrix operand order repaired and checked by reading, not compiled, not run on hardware
//
#include <hip/hip_runtime.h>
#include <hip/hip_bf16.h>

// ---------------------------------------------------------------------------
// EGNN forward for MI455X (gfx1250, wave32, WMMA f32_16x16x32_f16).
// - Edge/node MLPs as f16 WMMA, f32 accumulate; f16 shadow copy of h halves
//   the per-edge gather bytes (dominant HBM traffic at 23.3 TB/s).
// - Edge kernel batches 2 edge tiles per wave so every B-tile load feeds two
//   WMMAs (halves VMEM instructions in the hot loop).
// - Embed/node epilogues transpose D through LDS so global stores (and the
//   node residual read) are contiguous b128 ops instead of 64 scattered b32.
// - Segment sum via global_atomic_add_f32.
// ---------------------------------------------------------------------------

typedef __attribute__((ext_vector_type(16))) _Float16 v16h;
typedef __attribute__((ext_vector_type(8)))  float    v8f;

#define HID 128
#define WPB 4  // waves per block

static __device__ __forceinline__ v8f wmma16(v16h a, v16h b, v8f c) {
  // D = A(16x32,f16) * B(32x16,f16) + C(16x16,f32)
  return __builtin_amdgcn_wmma_f32_16x16x32_f16(
      /*neg_a=*/false, a, /*neg_b=*/false, b,
      /*c_mod=*/(short)0, c, /*reuse_a=*/false, /*reuse_b=*/false);
}

// B tile (32x16 f16) from pre-packed weights: per-lane 16 halves contiguous.
// Packed layout: P[((chunk*8 + ntile)*32 + lane)*16 + i]
static __device__ __forceinline__ v16h load_b_tile(const _Float16* __restrict__ P,
                                                   int c, int t, int lane) {
  const _Float16* p = P + (size_t)(((c * 8 + t) * 32 + lane) << 4);
  v16h r;
  __builtin_memcpy(&r, p, 32);  // 2x b128
  return r;
}

// A tile (16x32 f16) for one lane from a row-major f16 row.
// lane<16 (half=0): K = kb+[0..7], kb+[16..23]; lane>=16: K = kb+[8..15], kb+[24..31]
static __device__ __forceinline__ v16h load_a_row_f16(const _Float16* p, int kb, int half) {
  const _Float16* p0 = p + kb + half * 8;
  v16h r;
  __builtin_memcpy(&r, p0, 16);
  __builtin_memcpy(((_Float16*)&r) + 8, p0 + 16, 16);
  return r;
}

// Same but source row is f32 (converted in-register).
static __device__ __forceinline__ v16h load_a_row_f32(const float* p, int kb, int half) {
  const float* p0 = p + kb + half * 8;
  v16h r;
#pragma unroll
  for (int i = 0; i < 8; ++i) r[i] = (_Float16)p0[i];
#pragma unroll
  for (int i = 0; i < 8; ++i) r[i + 8] = (_Float16)p0[i + 16];
  return r;
}

// ---------------------------------------------------------------------------
// Weight packing: W [Kreal x 128] f32 row-major -> B-native f16 layout,
// zero-padded to chunks*32 rows. One thread per packed half.
// ---------------------------------------------------------------------------
__global__ void k_pack(const float* __restrict__ W, _Float16* __restrict__ P,
                       int Kreal, int total) {
  int idx = blockIdx.x * blockDim.x + threadIdx.x;
  if (idx >= total) return;
  int i    = idx & 15;
  int lane = (idx >> 4) & 31;
  int t    = (idx >> 9) & 7;
  int c    = idx >> 12;
  int col  = t * 16 + (lane & 15);
  int k    = c * 32 + ((lane >= 16) ? 8 : 0) + ((i < 8) ? i : i + 8);
  float v  = (k < Kreal) ? W[(size_t)k * HID + col] : 0.0f;
  P[idx] = (_Float16)v;
}

__global__ void k_zero(float* __restrict__ p, size_t n) {
  size_t i = (size_t)blockIdx.x * blockDim.x + threadIdx.x;
  size_t s = (size_t)gridDim.x * blockDim.x;
  for (; i < n; i += s) p[i] = 0.0f;
}

// ---------------------------------------------------------------------------
// Embedding: h = h_in[Nx32] @ w_in[32x128] + b_in.  K=32 = one WMMA chunk.
// Epilogue transposes through LDS for contiguous b128 stores.
// ---------------------------------------------------------------------------
__global__ void __launch_bounds__(WPB * 32) k_embed(
    const float* __restrict__ hin, const _Float16* __restrict__ p_win,
    const float* __restrict__ b_in,
    float* __restrict__ h32, _Float16* __restrict__ h16, int nTiles) {
  __shared__ float lds_o[WPB][16][HID];
  int wslot = threadIdx.x >> 5;
  int wave = blockIdx.x * WPB + wslot;
  if (wave >= nTiles) return;
  int lane = threadIdx.x & 31, half = lane >> 4, l16 = lane & 15;

  v16h a = load_a_row_f32(hin + (size_t)(wave * 16 + l16) * 32, 0, half);
  v8f zero = {};
#pragma unroll
  for (int t = 0; t < 8; ++t) {
    v16h b = load_b_tile(p_win, 0, t, lane);
    v8f acc = wmma16(a, b, zero);
    int col = t * 16 + l16;
    float bias = b_in[col];
#pragma unroll
    for (int r = 0; r < 8; ++r) lds_o[wslot][r + 8 * half][col] = acc[r] + bias;
  }

  // row phase: lane (l16, half) owns row l16, cols [half*64, half*64+64)
  int row = l16, colb = half * 64;
  size_t base = (size_t)(wave * 16 + row) * HID + colb;
  float* p32 = h32 + base;
  _Float16* p16 = h16 + base;
#pragma unroll
  for (int j = 0; j < 64; j += 8) {
    float o[8];
    __builtin_memcpy(o, &lds_o[wslot][row][colb + j], 32);
    __builtin_memcpy(p32 + j, o, 32);
    _Float16 q[8];
#pragma unroll
    for (int i = 0; i < 8; ++i) q[i] = (_Float16)o[i];
    __builtin_memcpy(p16 + j, q, 16);
  }
}

// ---------------------------------------------------------------------------
// Edge MLP + segment-sum.  One wave = 2 tiles of 16 edges (B tiles shared).
// GEMM1: A = [h[row] | h[col] | dist | 0-pad]  (K padded 257 -> 288, 9 chunks)
// GEMM2: K=128 (4 chunks), A re-laid out through LDS.
// ---------------------------------------------------------------------------
__global__ void __launch_bounds__(WPB * 32) k_edge(
    const _Float16* __restrict__ h16,
    const int* __restrict__ erow, const int* __restrict__ ecol,
    const float* __restrict__ dist, const float* __restrict__ emask,
    const _Float16* __restrict__ p_ew1, const float* __restrict__ eb1,
    const _Float16* __restrict__ p_ew2, const float* __restrict__ eb2,
    float* __restrict__ agg, int nTiles) {
  __shared__ _Float16 lds_e[WPB][2][16][HID];
  int wslot = threadIdx.x >> 5;
  int wave = blockIdx.x * WPB + wslot;
  if (wave * 2 >= nTiles) return;
  int lane = threadIdx.x & 31, half = lane >> 4, l16 = lane & 15;
  bool has1 = (wave * 2 + 1) < nTiles;  // wave-uniform

  const _Float16* prow[2];
  const _Float16* pcol[2];
  float dA[2];
  int ebase[2];
#pragma unroll
  for (int u = 0; u < 2; ++u) {
    int tile = wave * 2 + u;
    if (tile >= nTiles) tile = nTiles - 1;  // clamp; epilogue is guarded
    ebase[u] = tile * 16;
    int eA = ebase[u] + l16;  // both lane halves feed A-row M=l16
    prow[u] = h16 + (size_t)erow[eA] * HID;
    pcol[u] = h16 + (size_t)ecol[eA] * HID;
    dA[u] = dist[eA];
  }

  v8f zero = {};
  v8f acc[2][8];
#pragma unroll
  for (int u = 0; u < 2; ++u)
#pragma unroll
    for (int t = 0; t < 8; ++t) acc[u][t] = zero;

#pragma unroll
  for (int c = 0; c < 9; ++c) {
    v16h a[2];
#pragma unroll
    for (int u = 0; u < 2; ++u) {
      v16h av = {};
      if (c < 4)      av = load_a_row_f16(prow[u], c * 32, half);
      else if (c < 8) av = load_a_row_f16(pcol[u], (c - 4) * 32, half);
      else if (half == 0) av[0] = (_Float16)dA[u];  // K=256 = dist, rest pad
      a[u] = av;
    }
#pragma unroll
    for (int t = 0; t < 8; ++t) {
      v16h b = load_b_tile(p_ew1, c, t, lane);  // shared by both tiles
#pragma unroll
      for (int u = 0; u < 2; ++u) acc[u][t] = wmma16(a[u], b, acc[u][t]);
    }
  }

  // bias + ReLU -> LDS (f16) for GEMM2 A re-layout
#pragma unroll
  for (int u = 0; u < 2; ++u)
#pragma unroll
    for (int t = 0; t < 8; ++t) {
      int col = t * 16 + l16;
      float bias = eb1[col];
#pragma unroll
      for (int r = 0; r < 8; ++r) {
        float v = acc[u][t][r] + bias;
        lds_e[wslot][u][r + 8 * half][col] = (_Float16)(v > 0.0f ? v : 0.0f);
      }
    }

  v8f acc2[2][8];
#pragma unroll
  for (int u = 0; u < 2; ++u)
#pragma unroll
    for (int t = 0; t < 8; ++t) acc2[u][t] = zero;
#pragma unroll
  for (int c = 0; c < 4; ++c) {
    v16h a[2];
#pragma unroll
    for (int u = 0; u < 2; ++u)
      a[u] = load_a_row_f16(&lds_e[wslot][u][l16][0], c * 32, half);
#pragma unroll
    for (int t = 0; t < 8; ++t) {
      v16h b = load_b_tile(p_ew2, c, t, lane);
#pragma unroll
      for (int u = 0; u < 2; ++u) acc2[u][t] = wmma16(a[u], b, acc2[u][t]);
    }
  }

  // bias + ReLU + edge-mask, then f32 atomic segment-sum into agg[row]
#pragma unroll
  for (int u = 0; u < 2; ++u) {
    if (u == 1 && !has1) break;  // uniform guard: skip duplicate epilogue
    float msk[8];
    int   dstn[8];
#pragma unroll
    for (int r = 0; r < 8; ++r) {
      int e = ebase[u] + r + 8 * half;
      msk[r] = emask[e];
      dstn[r] = erow[e];
    }
#pragma unroll
    for (int t = 0; t < 8; ++t) {
      int col = t * 16 + l16;
      float bias = eb2[col];
#pragma unroll
      for (int r = 0; r < 8; ++r) {
        float v = acc2[u][t][r] + bias;
        v = (v > 0.0f ? v : 0.0f) * msk[r];
        atomicAdd(&agg[(size_t)dstn[r] * HID + col], v);
      }
    }
  }
}

// ---------------------------------------------------------------------------
// Node MLP: h = (h + relu([h|agg] @ nw1 + nb1) @ nw2 + nb2) * node_mask.
// In-place on h32/h16 (a wave only touches its own 16 rows). Epilogue goes
// through LDS so the residual read and both stores are contiguous b128 ops.
// ---------------------------------------------------------------------------
__global__ void __launch_bounds__(WPB * 32) k_node(
    float* __restrict__ h32, _Float16* __restrict__ h16,
    const float* __restrict__ agg, const float* __restrict__ nmask,
    const _Float16* __restrict__ p_nw1, const float* __restrict__ nb1,
    const _Float16* __restrict__ p_nw2, const float* __restrict__ nb2,
    int nTiles) {
  __shared__ _Float16 lds_m[WPB][16][HID];
  __shared__ float    lds_o[WPB][16][HID];
  int wslot = threadIdx.x >> 5;
  int wave = blockIdx.x * WPB + wslot;
  if (wave >= nTiles) return;
  int lane = threadIdx.x & 31, half = lane >> 4, l16 = lane & 15;
  int base = wave * 16;

  const _Float16* prow = h16 + (size_t)(base + l16) * HID;
  const float*    pagg = agg + (size_t)(base + l16) * HID;

  v8f zero = {};
  v8f acc[8];
#pragma unroll
  for (int t = 0; t < 8; ++t) acc[t] = zero;

#pragma unroll
  for (int c = 0; c < 8; ++c) {  // K=256: [h | agg]
    v16h a = (c < 4) ? load_a_row_f16(prow, c * 32, half)
                     : load_a_row_f32(pagg, (c - 4) * 32, half);
#pragma unroll
    for (int t = 0; t < 8; ++t) {
      v16h b = load_b_tile(p_nw1, c, t, lane);
      acc[t] = wmma16(a, b, acc[t]);
    }
  }

#pragma unroll
  for (int t = 0; t < 8; ++t) {
    int col = t * 16 + l16;
    float bias = nb1[col];
#pragma unroll
    for (int r = 0; r < 8; ++r) {
      float v = acc[t][r] + bias;
      lds_m[wslot][r + 8 * half][col] = (_Float16)(v > 0.0f ? v : 0.0f);
    }
  }

  v8f acc2[8];
#pragma unroll
  for (int t = 0; t < 8; ++t) acc2[t] = zero;
#pragma unroll
  for (int c = 0; c < 4; ++c) {
    v16h a = load_a_row_f16(&lds_m[wslot][l16][0], c * 32, half);
#pragma unroll
    for (int t = 0; t < 8; ++t) {
      v16h b = load_b_tile(p_nw2, c, t, lane);
      acc2[t] = wmma16(a, b, acc2[t]);
    }
  }

  // D + nb2 -> LDS (f32), then row phase: residual + mask with b128 traffic
#pragma unroll
  for (int t = 0; t < 8; ++t) {
    int col = t * 16 + l16;
    float bias = nb2[col];
#pragma unroll
    for (int r = 0; r < 8; ++r) lds_o[wslot][r + 8 * half][col] = acc2[t][r] + bias;
  }

  int row = l16, colb = half * 64;
  int node = base + row;
  float m = nmask[node];
  float* p32 = h32 + (size_t)node * HID + colb;
  _Float16* p16 = h16 + (size_t)node * HID + colb;
#pragma unroll
  for (int j = 0; j < 64; j += 8) {
    float dtmp[8], htmp[8], o[8];
    __builtin_memcpy(dtmp, &lds_o[wslot][row][colb + j], 32);
    __builtin_memcpy(htmp, p32 + j, 32);
#pragma unroll
    for (int i = 0; i < 8; ++i) o[i] = (htmp[i] + dtmp[i]) * m;
    __builtin_memcpy(p32 + j, o, 32);
    _Float16 q[8];
#pragma unroll
    for (int i = 0; i < 8; ++i) q[i] = (_Float16)o[i];
    __builtin_memcpy(p16 + j, q, 16);
  }
}

// ---------------------------------------------------------------------------
// Output head: out[Nx10] = h32 @ w_out[128x10] + b_out.  Tiny, VALU is fine.
// ---------------------------------------------------------------------------
__global__ void k_out(const float* __restrict__ h32, const float* __restrict__ w_out,
                      const float* __restrict__ b_out, float* __restrict__ out, int N) {
  int n = blockIdx.x * blockDim.x + threadIdx.x;
  if (n >= N) return;
  float acc[10];
#pragma unroll
  for (int j = 0; j < 10; ++j) acc[j] = b_out[j];
  const float* hp = h32 + (size_t)n * HID;
  for (int k = 0; k < HID; ++k) {
    float hv = hp[k];
#pragma unroll
    for (int j = 0; j < 10; ++j) acc[j] += hv * w_out[k * 10 + j];
  }
#pragma unroll
  for (int j = 0; j < 10; ++j) out[(size_t)n * 10 + j] = acc[j];
}

// ---------------------------------------------------------------------------
extern "C" void kernel_launch(void* const* d_in, const int* in_sizes, int n_in,
                              void* d_out, int out_size, void* d_ws, size_t ws_size,
                              hipStream_t stream) {
  (void)n_in; (void)out_size; (void)ws_size;
  const float* h_in  = (const float*)d_in[0];
  const int*   edges = (const int*)d_in[1];
  const float* dist  = (const float*)d_in[2];
  const float* nmask = (const float*)d_in[3];
  const float* emask = (const float*)d_in[4];
  const float* w_in  = (const float*)d_in[5];
  const float* b_in  = (const float*)d_in[6];
  const float* w_out = (const float*)d_in[7];
  const float* b_out = (const float*)d_in[8];
  const float* ew1   = (const float*)d_in[9];
  const float* eb1   = (const float*)d_in[10];
  const float* ew2   = (const float*)d_in[11];
  const float* eb2   = (const float*)d_in[12];
  const float* nw1   = (const float*)d_in[13];
  const float* nb1   = (const float*)d_in[14];
  const float* nw2   = (const float*)d_in[15];
  const float* nb2   = (const float*)d_in[16];

  const int N = in_sizes[3];  // node_mask has N elements
  const int E = in_sizes[4];  // edge_mask has E elements
  const int* erow = edges;
  const int* ecol = edges + E;

  // Workspace carve-up (~64.5 MB)
  char* ws = (char*)d_ws;
  size_t off = 0;
  auto take = [&](size_t bytes) -> char* {
    char* p = ws + off;
    off = (off + bytes + 255) & ~(size_t)255;
    return p;
  };
  float*    h32 = (float*)take((size_t)N * HID * 4);
  _Float16* h16 = (_Float16*)take((size_t)N * HID * 2);
  float*    agg = (float*)take((size_t)N * HID * 4);
  _Float16* p_win = (_Float16*)take((size_t)1 * 4096 * 2);
  _Float16 *p_ew1[2], *p_ew2[2], *p_nw1[2], *p_nw2[2];
  for (int l = 0; l < 2; ++l) {
    p_ew1[l] = (_Float16*)take((size_t)9 * 4096 * 2);
    p_ew2[l] = (_Float16*)take((size_t)4 * 4096 * 2);
    p_nw1[l] = (_Float16*)take((size_t)8 * 4096 * 2);
    p_nw2[l] = (_Float16*)take((size_t)4 * 4096 * 2);
  }

  auto pack = [&](const float* W, _Float16* P, int Kreal, int chunks) {
    int total = chunks * 4096;
    k_pack<<<(total + 255) / 256, 256, 0, stream>>>(W, P, Kreal, total);
  };
  pack(w_in, p_win, 32, 1);
  for (int l = 0; l < 2; ++l) {
    pack(ew1 + (size_t)l * 257 * HID, p_ew1[l], 257, 9);
    pack(ew2 + (size_t)l * HID * HID, p_ew2[l], 128, 4);
    pack(nw1 + (size_t)l * 256 * HID, p_nw1[l], 256, 8);
    pack(nw2 + (size_t)l * HID * HID, p_nw2[l], 128, 4);
  }

  const int nTilesN = N / 16;            // 3125
  const int nTilesE = E / 16;            // 50000
  const int nPairsE = (nTilesE + 1) / 2; // 25000 waves, 2 tiles each

  k_embed<<<(nTilesN + WPB - 1) / WPB, WPB * 32, 0, stream>>>(
      h_in, p_win, b_in, h32, h16, nTilesN);

  for (int l = 0; l < 2; ++l) {
    k_zero<<<4096, 256, 0, stream>>>(agg, (size_t)N * HID);
    k_edge<<<(nPairsE + WPB - 1) / WPB, WPB * 32, 0, stream>>>(
        h16, erow, ecol, dist, emask,
        p_ew1[l], eb1 + l * HID, p_ew2[l], eb2 + l * HID, agg, nTilesE);
    k_node<<<(nTilesN + WPB - 1) / WPB, WPB * 32, 0, stream>>>(
        h32, h16, agg, nmask,
        p_nw1[l], nb1 + l * HID, p_nw2[l], nb2 + l * HID, nTilesN);
  }

  k_out<<<(N + 255) / 256, 256, 0, stream>>>(h32, w_out, b_out, (float*)d_out, N);
}
